// SafetyReflexSNN_26070451487148
// MI455X (gfx1250) — matrix operand order repaired and checked by reading, not compiled, not run on hardware
//
#include <hip/hip_runtime.h>

typedef __bf16 bf16;
typedef __attribute__((ext_vector_type(16))) bf16  v16bf;
typedef __attribute__((ext_vector_type(8)))  bf16  v8bf;
typedef __attribute__((ext_vector_type(8)))  float v8f;
typedef __attribute__((ext_vector_type(4)))  float v4f;

#define NB_IN   512
#define NB_H0   1024
#define NB_H1   1024
#define NB_H2   1024
#define NB_H3   512
#define NB_OUT  128
#define BB      128
#define TT      512
#define KC      128          // K-chunk staged in LDS per block

// ---------------- WMMA helper (CDNA5 gfx1250, wave32) ----------------

__device__ __forceinline__ v8f wmma_bf16(v16bf a, v16bf b, v8f c) {
    return __builtin_amdgcn_wmma_f32_16x16x32_bf16(
        false, a, false, b, (short)0, c, false, false);
}

// A-matrix 16x32 bf16 fragment (ISA 7.12.2):
// lanes 0-15 : row M=lane,    VGPR0-3 = K=k0..k0+7,  VGPR4-7 = K=k0+16..k0+23
// lanes 16-31: row M=lane-16, VGPR0-3 = K=k0+8..+15, VGPR4-7 = K=k0+24..+31
__device__ __forceinline__ v16bf load_a_bf16(const bf16* __restrict__ A,
                                             int lda, int m0, int k0) {
    int lane = threadIdx.x & 31;
    int m  = m0 + (lane & 15);
    int kb = k0 + ((lane & 16) ? 8 : 0);
    const bf16* p = A + (size_t)m * lda + kb;
    v8bf lo = *(const v8bf*)(p);
    v8bf hi = *(const v8bf*)(p + 16);
    v16bf r;
#pragma unroll
    for (int i = 0; i < 8; ++i) { r[i] = lo[i]; r[i + 8] = hi[i]; }
    return r;
}

__device__ __forceinline__ v16bf load_a_f32(const float* __restrict__ A,
                                            int lda, int m0, int k0) {
    int lane = threadIdx.x & 31;
    int m  = m0 + (lane & 15);
    int kb = k0 + ((lane & 16) ? 8 : 0);
    const float* p = A + (size_t)m * lda + kb;
    v4f x0 = *(const v4f*)(p);
    v4f x1 = *(const v4f*)(p + 4);
    v4f x2 = *(const v4f*)(p + 16);
    v4f x3 = *(const v4f*)(p + 20);
    v16bf r;
#pragma unroll
    for (int i = 0; i < 4; ++i) {
        r[i]      = (bf16)x0[i];
        r[i + 4]  = (bf16)x1[i];
        r[i + 8]  = (bf16)x2[i];
        r[i + 12] = (bf16)x3[i];
    }
    return r;
}

// B-matrix 32x16 bf16 fragment served from the LDS-staged W tile.
// Tile layout: Wt[n_local * KC + k_local], n_local in [0,64), k_local in [0,KC).
__device__ __forceinline__ v16bf load_b_lds(const bf16* Wt, int nl0, int kk) {
    int lane = threadIdx.x & 31;
    int n  = nl0 + (lane & 15);
    int kb = kk + ((lane & 16) ? 8 : 0);
    const bf16* p = Wt + n * KC + kb;
    v8bf lo = *(const v8bf*)(p);       // ds_load_b128
    v8bf hi = *(const v8bf*)(p + 16);  // ds_load_b128
    v16bf r;
#pragma unroll
    for (int i = 0; i < 8; ++i) { r[i] = lo[i]; r[i + 8] = hi[i]; }
    return r;
}

// Issue one chunk stage: async-copy W[n0_blk..+63, k0..k0+KC) into LDS buffer.
// 256 threads x 4 x 16B = 16 KB, via GLOBAL_LOAD_ASYNC_TO_LDS_B128 (ASYNCcnt).
__device__ __forceinline__ void stage_w_async(const bf16* W, unsigned ldw,
                                              unsigned n0_blk, unsigned k0,
                                              bf16* buf, int tid) {
    unsigned ldsb = (unsigned)(uintptr_t)buf;           // low 32 bits = LDS offset
    unsigned long long base = (unsigned long long)(uintptr_t)W;
#pragma unroll
    for (int i = 0; i < 4; ++i) {
        unsigned o  = (unsigned)(i * 256 + tid) * 16u;  // byte offset within tile
        unsigned r  = o / (KC * 2u);                    // row (n) within tile
        unsigned cb = o % (KC * 2u);                    // byte col within row
        unsigned goff = ((n0_blk + r) * ldw + k0) * 2u + cb;
        asm volatile("global_load_async_to_lds_b128 %0, %1, %2"
                     :: "v"(ldsb + o), "v"(goff), "s"(base)
                     : "memory");
    }
}

// One GEMM pass: acc += A[m0:m0+16, :KDIM] * W[n0_blk:+64, :KDIM]^T
// W double-buffered through LDS with async prefetch of the next chunk.
template <bool A_F32, int KDIM>
__device__ __forceinline__ void gemm_pass(const void* __restrict__ Av, int lda,
                                          const bf16* __restrict__ W,
                                          bf16 (*smem)[64 * KC],
                                          v8f acc[4], int m0, int n0_blk, int tid) {
    constexpr int NC = KDIM / KC;
    stage_w_async(W, (unsigned)KDIM, (unsigned)n0_blk, 0u, smem[0], tid);
    for (int c = 0; c < NC; ++c) {
        const int cur = c & 1;
        if (c + 1 < NC) {
            stage_w_async(W, (unsigned)KDIM, (unsigned)n0_blk,
                          (unsigned)((c + 1) * KC), smem[cur ^ 1], tid);
            asm volatile("s_wait_asynccnt 0x4" ::: "memory");
        } else {
            asm volatile("s_wait_asynccnt 0x0" ::: "memory");
        }
        __syncthreads();
        const bf16* Wt = smem[cur];
        const int kbase = c * KC;
#pragma unroll
        for (int kk = 0; kk < KC; kk += 32) {
            v16bf a = A_F32 ? load_a_f32((const float*)Av, lda, m0, kbase + kk)
                            : load_a_bf16((const bf16*)Av, lda, m0, kbase + kk);
            v16bf b0 = load_b_lds(Wt,  0, kk);
            v16bf b1 = load_b_lds(Wt, 16, kk);
            v16bf b2 = load_b_lds(Wt, 32, kk);
            v16bf b3 = load_b_lds(Wt, 48, kk);
            acc[0] = wmma_bf16(a, b0, acc[0]);
            acc[1] = wmma_bf16(a, b1, acc[1]);
            acc[2] = wmma_bf16(a, b2, acc[2]);
            acc[3] = wmma_bf16(a, b3, acc[3]);
            // Pin the pipeline: all 8 DS reads issue before the 4 WMMAs, so
            // the wait inserter uses partial DScnt waits instead of draining
            // to zero before every WMMA. A-path (VMEM+cvt) floats freely.
            __builtin_amdgcn_sched_group_barrier(0x100, 8, 0); // DS read x8
            __builtin_amdgcn_sched_group_barrier(0x008, 4, 0); // WMMA    x4
        }
        __syncthreads();   // all waves done with smem[cur] before it is re-staged
    }
}

// ---------------- Fused  current = A0@W0^T (+ A1@W1^T)  ->  LIF ----------------
// Block = 256 threads = 8 waves; wave w computes rows [16w,16w+16) of the block's
// 64-column n-group; all 8 waves share one LDS-staged W tile. grid.x = N/64.

template <bool A_F32, int KDIM, bool DUAL, bool IS_OUT>
__global__ __launch_bounds__(256) void lif_gemm_kernel(
    const void* __restrict__ A0v, int lda0, const bf16* __restrict__ W0,
    const bf16* __restrict__ A1,  int lda1, const bf16* __restrict__ W1,
    float* __restrict__ Vst,            // [128, N] membrane state
    bf16*  __restrict__ Sout,           // [128, N] spikes (bf16) when !IS_OUT
    float* __restrict__ outSp, int ldo, // d_out spikes (f32) when IS_OUT
    float* __restrict__ counts,         // [128, N] spike counts when IS_OUT
    int N, float alpha, float vth) {

    __shared__ bf16 smem[2][64 * KC];   // 32 KB of the 320 KB WGP LDS

    const int tid  = threadIdx.x;
    const int lane = tid & 31;
    const int m0 = (tid >> 5) * 16;            // wave-in-block -> m tile
    const int n0 = blockIdx.x * 64;            // block -> n group

    v8f acc[4];
#pragma unroll
    for (int j = 0; j < 4; ++j) acc[j] = (v8f){0.f,0.f,0.f,0.f,0.f,0.f,0.f,0.f};

    gemm_pass<A_F32, KDIM>(A0v, lda0, W0, smem, acc, m0, n0, tid);
    if (DUAL)
        gemm_pass<false, KDIM>(A1, lda1, W1, smem, acc, m0, n0, tid);

    // LIF. C/D layout: lane -> col n0+(lane&15); VGPR i -> row m0+i(+8 if lane>=16)
    const int n_base = n0 + (lane & 15);
    const int m_base = m0 + ((lane & 16) ? 8 : 0);
#pragma unroll
    for (int j = 0; j < 4; ++j) {
        const int n = n_base + j * 16;
#pragma unroll
        for (int i = 0; i < 8; ++i) {
            const int m = m_base + i;
            const size_t vi = (size_t)m * N + n;
            float v = alpha * Vst[vi] + acc[j][i];
            float s = (v >= vth) ? 1.0f : 0.0f;
            Vst[vi] = v * (1.0f - s);
            if (IS_OUT) {
                outSp[(size_t)m * ldo + n] = s;
                counts[(size_t)m * N + n] += s;
            } else {
                Sout[(size_t)m * N + n] = (bf16)s;
            }
        }
    }
}

// ---------------- prep kernels ----------------

__global__ void cvt_f32_to_bf16(const float* __restrict__ in,
                                bf16* __restrict__ out, int n) {
    int i = blockIdx.x * blockDim.x + threadIdx.x;
    if (i < n) out[i] = (bf16)in[i];
}

__global__ void zero_f32_k(float* __restrict__ p, int n) {
    int i = blockIdx.x * blockDim.x + threadIdx.x;
    if (i < n) p[i] = 0.0f;
}

// ---------------- host launcher ----------------

extern "C" void kernel_launch(void* const* d_in, const int* in_sizes, int n_in,
                              void* d_out, int out_size, void* d_ws, size_t ws_size,
                              hipStream_t stream) {
    (void)in_sizes; (void)n_in; (void)out_size; (void)ws_size;

    const float* x = (const float*)d_in[0];              // [B,T,N_IN] f32
    const float* Wf[6] = {
        (const float*)d_in[1],  // W_in_h0 [1024,512]
        (const float*)d_in[2],  // W_h0_h1 [1024,1024]
        (const float*)d_in[3],  // W_h1_h2 [1024,1024]
        (const float*)d_in[4],  // W_h2_h2 [1024,1024]
        (const float*)d_in[5],  // W_h2_h3 [512,1024]
        (const float*)d_in[6],  // W_h3_out [128,512]
    };
    const int wsz[6] = { NB_H0*NB_IN, NB_H1*NB_H0, NB_H2*NB_H1,
                         NB_H2*NB_H2, NB_H3*NB_H2, NB_OUT*NB_H3 };

    char* wp = (char*)d_ws;
    auto bump = [&](size_t bytes) -> char* {
        char* p = wp; wp += (bytes + 255) & ~(size_t)255; return p;
    };
    bf16* Wb[6];
    for (int i = 0; i < 6; ++i) Wb[i] = (bf16*)bump((size_t)wsz[i] * sizeof(bf16));

    const int nVf = BB * (NB_H0 + NB_H1 + NB_H2 + NB_H3 + NB_OUT);
    float* Vall = (float*)bump((size_t)nVf * sizeof(float));
    float* V0 = Vall;
    float* V1 = V0 + (size_t)BB * NB_H0;
    float* V2 = V1 + (size_t)BB * NB_H1;
    float* V3 = V2 + (size_t)BB * NB_H2;
    float* V4 = V3 + (size_t)BB * NB_H3;

    bf16* S0  = (bf16*)bump((size_t)BB * NB_H0 * sizeof(bf16));
    bf16* S1  = (bf16*)bump((size_t)BB * NB_H1 * sizeof(bf16));
    bf16* S2a = (bf16*)bump((size_t)BB * NB_H2 * sizeof(bf16));
    bf16* S2b = (bf16*)bump((size_t)BB * NB_H2 * sizeof(bf16));
    bf16* S3  = (bf16*)bump((size_t)BB * NB_H3 * sizeof(bf16));

    float* outSp  = (float*)d_out;                          // [B,T,N_OUT]
    float* counts = outSp + (size_t)BB * TT * NB_OUT;       // [B,N_OUT]

    for (int i = 0; i < 6; ++i)
        cvt_f32_to_bf16<<<(wsz[i] + 255) / 256, 256, 0, stream>>>(Wf[i], Wb[i], wsz[i]);

    zero_f32_k<<<(nVf + 255) / 256, 256, 0, stream>>>(Vall, nVf);
    const int s2f = (BB * NB_H2 * (int)sizeof(bf16)) / 4;
    zero_f32_k<<<(s2f + 255) / 256, 256, 0, stream>>>((float*)S2a, s2f);
    zero_f32_k<<<(s2f + 255) / 256, 256, 0, stream>>>((float*)S2b, s2f);
    zero_f32_k<<<(BB * NB_OUT + 255) / 256, 256, 0, stream>>>(counts, BB * NB_OUT);

    for (int t = 0; t < TT; ++t) {
        const float* xt = x + (size_t)t * NB_IN;
        bf16* s2prev = (t & 1) ? S2b : S2a;
        bf16* s2cur  = (t & 1) ? S2a : S2b;

        // h0: x_t @ W0^T  (A in f32, K=512)
        lif_gemm_kernel<true, NB_IN, false, false><<<NB_H0 / 64, 256, 0, stream>>>(
            xt, TT * NB_IN, Wb[0], nullptr, 0, nullptr,
            V0, S0, nullptr, 0, nullptr, NB_H0, 0.9f, 0.5f);

        // h1: s0 @ W1^T  (K=1024)
        lif_gemm_kernel<false, NB_H0, false, false><<<NB_H1 / 64, 256, 0, stream>>>(
            S0, NB_H0, Wb[1], nullptr, 0, nullptr,
            V1, S1, nullptr, 0, nullptr, NB_H1, 0.95f, 0.5f);

        // h2: s1 @ W2^T + s2_prev @ W2r^T  (dual, K=1024 each)
        lif_gemm_kernel<false, NB_H1, true, false><<<NB_H2 / 64, 256, 0, stream>>>(
            S1, NB_H1, Wb[2], s2prev, NB_H2, Wb[3],
            V2, s2cur, nullptr, 0, nullptr, NB_H2, 0.93f, 0.5f);

        // h3: s2 @ W3^T  (K=1024)
        lif_gemm_kernel<false, NB_H2, false, false><<<NB_H3 / 64, 256, 0, stream>>>(
            s2cur, NB_H2, Wb[4], nullptr, 0, nullptr,
            V3, S3, nullptr, 0, nullptr, NB_H3, 0.9f, 0.5f);

        // out: s3 @ W4^T  (K=512) -> f32 spikes into d_out + counts
        lif_gemm_kernel<false, NB_H3, false, true><<<NB_OUT / 64, 256, 0, stream>>>(
            S3, NB_H3, Wb[5], nullptr, 0, nullptr,
            V4, nullptr, outSp + (size_t)t * NB_OUT, TT * NB_OUT, counts,
            NB_OUT, 0.82f, 0.8f);
    }
}
